// MultiHeadAttention_15874199126440
// MI455X (gfx1250) — matrix operand order repaired and checked by reading, not compiled, not run on hardware
//
#include <hip/hip_runtime.h>
#include <hip/hip_bf16.h>

// Problem constants (from reference): B=4, S=2048, E=1024, H=16, HD=64
#define B_  4
#define S_  2048
#define E_  1024
#define H_  16
#define HD_ 64
#define F_  192            // 3*HD per head
#define SW_ (S_ / 32)      // mask words per row

typedef __attribute__((ext_vector_type(16))) __bf16 v16bf;
typedef __attribute__((ext_vector_type(8)))  float  v8f;

// ---------- helpers ----------

__device__ __forceinline__ unsigned short f2bf(float f) {
  unsigned int x = __float_as_uint(f);
  x += 0x7FFFu + ((x >> 16) & 1u);          // round-to-nearest-even
  return (unsigned short)(x >> 16);
}

// A-matrix (16xK=32, bf16) per CDNA5 ISA layout:
// lanes 0-15: M=lane, K = {0..7, 16..23}; lanes 16-31: M=lane-16, K = {8..15, 24..31}
__device__ __forceinline__ v16bf load_a_tile(const unsigned short* base, int ld,
                                             int row0, int col0, int lane) {
  int mr = lane & 15, hi = lane >> 4;
  const unsigned short* p = base + (long long)(row0 + mr) * ld + col0 + hi * 8;
  union { uint4 q[2]; v16bf v; } t;
  t.q[0] = *reinterpret_cast<const uint4*>(p);        // K = hi*8 .. hi*8+7
  t.q[1] = *reinterpret_cast<const uint4*>(p + 16);   // K = 16+hi*8 .. 16+hi*8+7
  return t.v;
}

// B-matrix (K=32 x N=16, bf16): B(kk,n) = M[(n0+n)*ld + k0+kk]  (tile of M^T, M row-major)
// lanes 0-15: N=lane, K = {0..7, 16..23}; lanes 16-31: N=lane-16, K = {8..15, 24..31}
__device__ __forceinline__ v16bf load_bT_tile(const unsigned short* base, int ld,
                                              int n0, int k0, int lane) {
  int n = lane & 15, hi = lane >> 4;
  const unsigned short* p = base + (long long)(n0 + n) * ld + k0 + hi * 8;
  union { uint4 q[2]; v16bf v; } t;
  t.q[0] = *reinterpret_cast<const uint4*>(p);
  t.q[1] = *reinterpret_cast<const uint4*>(p + 16);
  return t.v;
}

__device__ __forceinline__ v8f wmma_bf16(v16bf a, v16bf b, v8f c) {
  return __builtin_amdgcn_wmma_f32_16x16x32_bf16(false, a, false, b, (short)0, c,
                                                 false, false);
}

// ---------- kernel 1: fp32 -> bf16 conversion ----------

__global__ void cvt_bf16_kernel(const float* __restrict__ src,
                                unsigned short* __restrict__ dst, int n) {
  int i = blockIdx.x * blockDim.x + threadIdx.x;
  if (i < n) dst[i] = f2bf(src[i]);
}

// ---------- kernel 1b: pack int32 mask -> bit mask (1 bit per column) ----------
// word w covers mask elements [w*32 .. w*32+31]; bit L set iff mask==1 at column L.
__global__ void pack_mask_kernel(const int* __restrict__ mask,
                                 unsigned int* __restrict__ bits, long long nwords) {
  long long w = ((long long)blockIdx.x * blockDim.x + threadIdx.x) >> 5;
  int lane = threadIdx.x & 31;
  if (w < nwords) {
    int v = mask[w * 32 + lane];
    unsigned long long bal = __ballot(v == 1);     // wave32: low 32 bits valid
    if (lane == 0) bits[w] = (unsigned int)bal;
  }
}

// ---------- kernel 2: fused per-head QKV projection, 2x2 register tiling ----------
// qkv[b,h,s,f] = sum_e x[b,s,e] * Wqkv[h,f,e] + bqkv[h,f]
// writes q,k row-major [BH, S, 64] and v transposed [BH, 64, S], all bf16.

__device__ __forceinline__ void store_qkv_tile(v8f acc, int bh, int h, int s0, int f0,
                                               const float* __restrict__ bias,
                                               unsigned short* __restrict__ qb,
                                               unsigned short* __restrict__ kb,
                                               unsigned short* __restrict__ vTb,
                                               int lane) {
  int n = lane & 15, hi = lane >> 4;
  float bv = bias[h * F_ + f0 + n];
  if (f0 < 64) {                                   // Q: [BH,S,64]
#pragma unroll
    for (int r = 0; r < 8; ++r)
      qb[((long long)bh * S_ + s0 + r + 8 * hi) * HD_ + f0 + n] = f2bf(acc[r] + bv);
  } else if (f0 < 128) {                           // K: [BH,S,64]
#pragma unroll
    for (int r = 0; r < 8; ++r)
      kb[((long long)bh * S_ + s0 + r + 8 * hi) * HD_ + (f0 - 64) + n] = f2bf(acc[r] + bv);
  } else {                                         // V^T: [BH,64,S] (8 contig bf16/lane)
    union { unsigned short u[8]; uint4 q; } pk;
#pragma unroll
    for (int r = 0; r < 8; ++r) pk.u[r] = f2bf(acc[r] + bv);
    long long off = ((long long)bh * HD_ + (f0 - 128) + n) * S_ + s0 + 8 * hi;
    *reinterpret_cast<uint4*>(vTb + off) = pk.q;
  }
}

__global__ __launch_bounds__(128) void qkv_kernel(
    const unsigned short* __restrict__ xb, const unsigned short* __restrict__ Wb,
    const float* __restrict__ bias, unsigned short* __restrict__ qb,
    unsigned short* __restrict__ kb, unsigned short* __restrict__ vTb) {
  int lane = threadIdx.x & 31, wave = threadIdx.x >> 5;
  int gid = blockIdx.x * 4 + wave;               // 64 bh * 64 s_st * 6 f_st = 24576 waves
  int f_st = gid % 6;
  int s_st = (gid / 6) & 63;
  int bh   = gid / (6 * 64);
  int b = bh >> 4, h = bh & 15;
  int s0 = s_st * 32, f0 = f_st * 32;            // 32x32 output supertile per wave

  const unsigned short* A = xb + (long long)b * S_ * E_;
  const unsigned short* W = Wb + (long long)h * F_ * E_;

  v8f a00 = {}, a01 = {}, a10 = {}, a11 = {};
  for (int e0 = 0; e0 < E_; e0 += 32) {
    v16bf x0 = load_a_tile(A, E_, s0,      e0, lane);
    v16bf x1 = load_a_tile(A, E_, s0 + 16, e0, lane);
    v16bf w0 = load_bT_tile(W, E_, f0,      e0, lane);
    v16bf w1 = load_bT_tile(W, E_, f0 + 16, e0, lane);
    a00 = wmma_bf16(x0, w0, a00);
    a01 = wmma_bf16(x0, w1, a01);
    a10 = wmma_bf16(x1, w0, a10);
    a11 = wmma_bf16(x1, w1, a11);
  }
  store_qkv_tile(a00, bh, h, s0,      f0,      bias, qb, kb, vTb, lane);
  store_qkv_tile(a01, bh, h, s0,      f0 + 16, bias, qb, kb, vTb, lane);
  store_qkv_tile(a10, bh, h, s0 + 16, f0,      bias, qb, kb, vTb, lane);
  store_qkv_tile(a11, bh, h, s0 + 16, f0 + 16, bias, qb, kb, vTb, lane);
}

// ---------- kernel 3: flash attention with mask==1 -> score=0 ----------
__global__ __launch_bounds__(128) void attn_kernel(
    const unsigned short* __restrict__ qb, const unsigned short* __restrict__ kb,
    const unsigned short* __restrict__ vTb, const unsigned int* __restrict__ maskbits,
    unsigned short* __restrict__ ctxb) {
  __shared__ unsigned short pt[4][16 * 32];        // per-wave P bounce buffer
  int lane = threadIdx.x & 31, wave = threadIdx.x >> 5;
  int gid = blockIdx.x * 4 + wave;                 // 8192 waves total
  int q_t = gid & 127;
  int bh  = gid >> 7;
  int b = bh >> 4, h = bh & 15;
  int q0 = q_t * 16;
  int n = lane & 15, hi = lane >> 4;

  const unsigned short* Q  = qb  + (long long)bh * S_ * HD_;
  const unsigned short* K  = kb  + (long long)bh * S_ * HD_;
  const unsigned short* VT = vTb + (long long)bh * HD_ * S_;
  const unsigned int* MW = maskbits + (long long)b * S_ * SW_;
  unsigned short* myp = pt[wave];

  v16bf aq0 = load_a_tile(Q, HD_, q0, 0, lane);
  v16bf aq1 = load_a_tile(Q, HD_, q0, 32, lane);

  float mrow[8], lrow[8];
#pragma unroll
  for (int r = 0; r < 8; ++r) { mrow[r] = -1e30f; lrow[r] = 0.f; }
  v8f acc0 = {}, acc1 = {}, acc2 = {}, acc3 = {};
  const float scale = 0.125f;                      // 1/sqrt(64)

  for (int j0 = 0; j0 < S_; j0 += 32) {
    // scores: two 16x16 tiles over 32 keys (K-dim = HD = 64 -> 2 wmma each)
    v8f sc0 = {}, sc1 = {};
    sc0 = wmma_bf16(aq0, load_bT_tile(K, HD_, j0,      0,  lane), sc0);
    sc0 = wmma_bf16(aq1, load_bT_tile(K, HD_, j0,      32, lane), sc0);
    sc1 = wmma_bf16(aq0, load_bT_tile(K, HD_, j0 + 16, 0,  lane), sc1);
    sc1 = wmma_bf16(aq1, load_bT_tile(K, HD_, j0 + 16, 32, lane), sc1);

    // scale + mask via packed bits (bit c of word = column j0+c); L2-resident (2 MB)
#pragma unroll
    for (int r = 0; r < 8; ++r) {
      unsigned int mw = MW[(long long)(q0 + r + 8 * hi) * SW_ + (j0 >> 5)];
      float s0v = sc0[r] * scale; if ((mw >> n) & 1u)        s0v = 0.f;
      float s1v = sc1[r] * scale; if ((mw >> (16 + n)) & 1u) s1v = 0.f;
      sc0[r] = s0v; sc1[r] = s1v;
    }

    // online softmax; D-layout row m = r + 8*hi lives on one 16-lane half
#pragma unroll
    for (int r = 0; r < 8; ++r) {
      float rm = fmaxf(sc0[r], sc1[r]);
      for (int off = 8; off; off >>= 1) rm = fmaxf(rm, __shfl_xor(rm, off, 16));
      float mn = fmaxf(mrow[r], rm);
      float alpha = __expf(mrow[r] - mn);
      float p0 = __expf(sc0[r] - mn);
      float p1 = __expf(sc1[r] - mn);
      float rs = p0 + p1;
      for (int off = 8; off; off >>= 1) rs += __shfl_xor(rs, off, 16);
      lrow[r] = lrow[r] * alpha + rs;
      mrow[r] = mn;
      acc0[r] *= alpha; acc1[r] *= alpha; acc2[r] *= alpha; acc3[r] *= alpha;
      int m = r + 8 * hi;
      myp[m * 32 + n]      = f2bf(p0);
      myp[m * 32 + 16 + n] = f2bf(p1);
    }

    // re-read P in A layout (intra-wave LDS, DS ops are in-order)
    v16bf ap = load_a_tile(myp, 32, 0, 0, lane);
    acc0 = wmma_bf16(ap, load_bT_tile(VT, S_, 0,  j0, lane), acc0);
    acc1 = wmma_bf16(ap, load_bT_tile(VT, S_, 16, j0, lane), acc1);
    acc2 = wmma_bf16(ap, load_bT_tile(VT, S_, 32, j0, lane), acc2);
    acc3 = wmma_bf16(ap, load_bT_tile(VT, S_, 48, j0, lane), acc3);
  }

  // normalize + store ctx (bf16, [B,S,E] with heads concatenated)
#pragma unroll
  for (int r = 0; r < 8; ++r) {
    float inv = 1.0f / lrow[r];
    long long ro = ((long long)b * S_ + q0 + r + 8 * hi) * E_ + h * HD_;
    ctxb[ro + n]      = f2bf(acc0[r] * inv);
    ctxb[ro + 16 + n] = f2bf(acc1[r] * inv);
    ctxb[ro + 32 + n] = f2bf(acc2[r] * inv);
    ctxb[ro + 48 + n] = f2bf(acc3[r] * inv);
  }
}

// ---------- kernel 4: output projection out = ctx @ Wo^T + bo, 2x2 tiling ----------
__global__ __launch_bounds__(128) void oproj_kernel(
    const unsigned short* __restrict__ ctxb, const unsigned short* __restrict__ Wob,
    const float* __restrict__ bo, float* __restrict__ out) {
  int lane = threadIdx.x & 31, wave = threadIdx.x >> 5;
  int gid = blockIdx.x * 4 + wave;                 // 256 m_st * 32 n_st = 8192 waves
  int n_st = gid & 31;
  int m_st = gid >> 5;
  int m0 = m_st * 32, n0 = n_st * 32;              // 32x32 output supertile per wave

  v8f a00 = {}, a01 = {}, a10 = {}, a11 = {};
  for (int c0 = 0; c0 < E_; c0 += 32) {
    v16bf x0 = load_a_tile(ctxb, E_, m0,      c0, lane);
    v16bf x1 = load_a_tile(ctxb, E_, m0 + 16, c0, lane);
    v16bf w0 = load_bT_tile(Wob, E_, n0,      c0, lane);
    v16bf w1 = load_bT_tile(Wob, E_, n0 + 16, c0, lane);
    a00 = wmma_bf16(x0, w0, a00);
    a01 = wmma_bf16(x0, w1, a01);
    a10 = wmma_bf16(x1, w0, a10);
    a11 = wmma_bf16(x1, w1, a11);
  }
  int n = lane & 15, hi = lane >> 4;
  float bv0 = bo[n0 + n], bv1 = bo[n0 + 16 + n];
#pragma unroll
  for (int r = 0; r < 8; ++r) {
    long long ro0 = (long long)(m0 + r + 8 * hi) * E_;
    long long ro1 = (long long)(m0 + 16 + r + 8 * hi) * E_;
    out[ro0 + n0 + n]      = a00[r] + bv0;
    out[ro0 + n0 + 16 + n] = a01[r] + bv1;
    out[ro1 + n0 + n]      = a10[r] + bv0;
    out[ro1 + n0 + 16 + n] = a11[r] + bv1;
  }
}

// ---------- launcher ----------

extern "C" void kernel_launch(void* const* d_in, const int* in_sizes, int n_in,
                              void* d_out, int out_size, void* d_ws, size_t ws_size,
                              hipStream_t stream) {
  const float* x    = (const float*)d_in[0];
  const int*   mask = (const int*)d_in[1];
  const float* Wqkv = (const float*)d_in[2];
  const float* bqkv = (const float*)d_in[3];
  const float* Wo   = (const float*)d_in[4];
  const float* bo   = (const float*)d_in[5];
  float* out = (float*)d_out;

  // workspace carve-up (bf16 elements)
  unsigned short* p = (unsigned short*)d_ws;
  unsigned short* xb    = p; p += (size_t)B_ * S_ * E_;        // 8.4M  (reused as ctx)
  unsigned short* Wqkvb = p; p += (size_t)H_ * F_ * E_;        // 3.1M
  unsigned short* Wob   = p; p += (size_t)E_ * E_;             // 1.0M
  unsigned short* qb    = p; p += (size_t)B_ * H_ * S_ * HD_;  // 8.4M
  unsigned short* kb    = p; p += (size_t)B_ * H_ * S_ * HD_;  // 8.4M
  unsigned short* vTb   = p; p += (size_t)B_ * H_ * S_ * HD_;  // 8.4M
  unsigned int*   mbits = (unsigned int*)p;                    // 524288 u32 = 2 MB
  unsigned short* ctxb  = xb;  // x is dead after the QKV GEMM; alias to save 16 MB

  int nx = B_ * S_ * E_;
  cvt_bf16_kernel<<<(nx + 255) / 256, 256, 0, stream>>>(x, xb, nx);
  int nw = H_ * F_ * E_;
  cvt_bf16_kernel<<<(nw + 255) / 256, 256, 0, stream>>>(Wqkv, Wqkvb, nw);
  int no = E_ * E_;
  cvt_bf16_kernel<<<(no + 255) / 256, 256, 0, stream>>>(Wo, Wob, no);
  long long nwords = (long long)B_ * S_ * SW_;                 // 524288
  pack_mask_kernel<<<(unsigned)((nwords * 32 + 255) / 256), 256, 0, stream>>>(mask, mbits, nwords);

  // 64 bh * 64 s-supertiles * 6 f-supertiles = 24576 waves / 4 per block
  qkv_kernel<<<6144, 128, 0, stream>>>(xb, Wqkvb, bqkv, qb, kb, vTb);
  // 64 bh * 128 q-tiles = 8192 waves / 4 per block
  attn_kernel<<<2048, 128, 0, stream>>>(qb, kb, vTb, mbits, ctxb);
  // 256 row-supertiles * 32 col-supertiles = 8192 waves / 4 per block
  oproj_kernel<<<2048, 128, 0, stream>>>(ctxb, Wob, bo, out);
}
